// AttentiveGraphPooling_49546742726912
// MI455X (gfx1250) — matrix operand order repaired and checked by reading, not compiled, not run on hardware
//
#include <hip/hip_runtime.h>
#include <stdint.h>

// Problem constants (match reference)
#define H     256
#define GNUM  4096

typedef float        f32x8  __attribute__((ext_vector_type(8)));
typedef __bf16       bf16x16 __attribute__((ext_vector_type(16)));
typedef unsigned int u32x8  __attribute__((ext_vector_type(8)));

__device__ __forceinline__ unsigned int pack2bf(float a, float b) {
    unsigned int ua = __builtin_bit_cast(unsigned int, a);
    unsigned int ub = __builtin_bit_cast(unsigned int, b);
    ua += 0x7FFFu + ((ua >> 16) & 1u);   // round-to-nearest-even
    ub += 0x7FFFu + ((ub >> 16) & 1u);
    return (ua >> 16) | (ub & 0xFFFF0000u);
}

__device__ __forceinline__ f32x8 wmma_bf16(u32x8 a, u32x8 b, f32x8 c) {
    return __builtin_amdgcn_wmma_f32_16x16x32_bf16(
        false, __builtin_bit_cast(bf16x16, a),
        false, __builtin_bit_cast(bf16x16, b),
        (short)0, c, false, false);
}

__device__ __forceinline__ float sigmoidf_(float x) {
    return 1.0f / (1.0f + expf(-x));
}

// A-matrix (16x32 bf16) LDS fragment load.
// Lane m (=lane&15) is row M; half (=lane>>4) selects K-interleave per ISA:
//  v<4 : K = 2v + 8*half ; v>=4 : K = 16 + 2(v-4) + 8*half  (pairs packed)
// LDS layout: [rows][128 packed-uint cols], so k/2 index:
//  v<4 -> c*16 + half*4 + v ; v>=4 -> c*16 + 8 + half*4 + (v-4)  => two uint4.
__device__ __forceinline__ u32x8 loadA_lds(const unsigned int* lds, int m, int half, int c) {
    const unsigned int* p = lds + m * 128 + c * 16 + half * 4;
    uint4 lo = *(const uint4*)p;
    uint4 hi = *(const uint4*)(p + 8);
    u32x8 r;
    r[0] = lo.x; r[1] = lo.y; r[2] = lo.z; r[3] = lo.w;
    r[4] = hi.x; r[5] = hi.y; r[6] = hi.z; r[7] = hi.w;
    return r;
}

// B fragments are pre-packed in global memory as frag[ot][c][lane][v] (uints),
// so each lane reads 8 contiguous dwords (two b128 loads, wave reads 1KB contiguous).
__device__ __forceinline__ u32x8 loadB_frag(const unsigned int* __restrict__ frag,
                                            int ot, int c, int lane) {
    const uint4* p = (const uint4*)(frag + ((size_t)((ot * 8 + c) * 32 + lane)) * 8);
    uint4 lo = p[0];
    uint4 hi = p[1];
    u32x8 r;
    r[0] = lo.x; r[1] = lo.y; r[2] = lo.z; r[3] = lo.w;
    r[4] = hi.x; r[5] = hi.y; r[6] = hi.z; r[7] = hi.w;
    return r;
}

// ---------------------------------------------------------------------------
// Pre-pack a row-major fp32 weight W (out_dim x 256), used as B = W^T
// (K=256 rows, N=out_dim cols), into bf16 WMMA B-fragments.
// ISA B 32x16 bf16 layout: lane = N (0..15) + 16*((K/8)&1); VGPR v packs a K pair (K, K+1) as below.
// frag element index e = (((ot*8 + c)*32 + lane)*8 + v)
//   o = ot*16 + (lane&15) ; k = c*32 + (lane>>4)*8 + (v<4 ? 2v : 16 + 2(v-4))
// ---------------------------------------------------------------------------
__global__ void __launch_bounds__(256) make_bfrag_kernel(const float* __restrict__ W,
                                                         unsigned int* __restrict__ frag,
                                                         int out_tiles) {
    int e = blockIdx.x * blockDim.x + threadIdx.x;
    int total = out_tiles * 8 * 32 * 8;
    if (e >= total) return;
    int v    = e & 7;
    int lane = (e >> 3) & 31;
    int c    = (e >> 8) & 7;
    int ot   = e >> 11;
    int o = ot * 16 + (lane & 15);
    int k = c * 32 + (lane >> 4) * 8 + (v < 4 ? v * 2 : 16 + (v - 4) * 2);
    frag[e] = pack2bf(W[(size_t)o * H + k], W[(size_t)o * H + k + 1]);
}

// ---------------------------------------------------------------------------
// Segment mean-pool over sorted `batch` (no atomics): one block per graph,
// binary search the node range, thread t reduces column t.
// useGate=0 -> plain mean of x; useGate=1 -> mean of x*gate[n].
// ---------------------------------------------------------------------------
__global__ void __launch_bounds__(256) pool_kernel(const float* __restrict__ x,
                                                   const float* __restrict__ gate,
                                                   const int* __restrict__ batch,
                                                   float* __restrict__ out,
                                                   int N, int useGate) {
    __shared__ int s_lo, s_hi;
    int g = blockIdx.x;
    if (threadIdx.x == 0) {
        int lo = 0, hi = N;
        while (lo < hi) { int m = (lo + hi) >> 1; if (batch[m] < g) lo = m + 1; else hi = m; }
        s_lo = lo;
        int lo2 = lo, hi2 = N;
        while (lo2 < hi2) { int m = (lo2 + hi2) >> 1; if (batch[m] < g + 1) lo2 = m + 1; else hi2 = m; }
        s_hi = lo2;
    }
    __syncthreads();
    int start = s_lo, end = s_hi;
    int cnt = end - start;
    float denom = (float)(cnt > 0 ? cnt : 1);
    int t = threadIdx.x;
    float sum = 0.0f;
    if (useGate) {
        for (int n = start; n < end; ++n)
            sum += x[(size_t)n * H + t] * gate[n];
    } else {
        for (int n = start; n < end; ++n)
            sum += x[(size_t)n * H + t];
    }
    out[(size_t)g * H + t] = sum / denom;
}

// ---------------------------------------------------------------------------
// Gate MLP, fully fused, TWO 16-node tiles per wave (1 wave / block):
//   gate_input = x + repr[batch]  (staged bf16 in LDS)
//   hgate = relu(gate_input @ W1^T + b1)         [32x256, WMMA bf16]
//   gate  = sigmoid(hgate @ w2 + b2)             [fused reduction]
// Each B fragment is loaded once and feeds two independent WMMAs; B loads are
// software-pipelined (prefetch chunk c+1 before the WMMAs of chunk c).
// ---------------------------------------------------------------------------
__global__ void __launch_bounds__(32) gate_kernel(const float* __restrict__ x,
                                                  const float* __restrict__ repr,
                                                  const int* __restrict__ batch,
                                                  const unsigned int* __restrict__ w1frag,
                                                  const float* __restrict__ b1,
                                                  const float* __restrict__ w2,
                                                  const float* __restrict__ b2,
                                                  float* __restrict__ gate,
                                                  int N) {
    __shared__ unsigned int ldsA[32 * 128];  // 32 rows x 256 bf16 (packed pairs)
    const int lane  = threadIdx.x;
    const int base  = blockIdx.x * 32;
    const int half  = lane >> 4;
    const int col16 = lane & 15;

    // Stage gate_input rows (fp32 add -> bf16 pack -> LDS)
    for (int i = 0; i < 32; ++i) {
        int node = base + i;
        if (node >= N) node = N - 1;
        int g = batch[node];
        const float4* xr = (const float4*)(x + (size_t)node * H);
        const float4* rr = (const float4*)(repr + (size_t)g * H);
#pragma unroll
        for (int j = 0; j < 2; ++j) {
            int f = lane + j * 32;         // float4 index 0..63
            float4 a = xr[f];
            float4 b = rr[f];
            ldsA[i * 128 + f * 2]     = pack2bf(a.x + b.x, a.y + b.y);
            ldsA[i * 128 + f * 2 + 1] = pack2bf(a.z + b.z, a.w + b.w);
        }
    }
    __syncthreads();

    // A fragments resident in registers for both row tiles (8 K-chunks each)
    u32x8 afrag0[8], afrag1[8];
#pragma unroll
    for (int c = 0; c < 8; ++c) {
        afrag0[c] = loadA_lds(ldsA,            col16, half, c);
        afrag1[c] = loadA_lds(ldsA + 16 * 128, col16, half, c);
    }

    float partial0[8], partial1[8];
#pragma unroll
    for (int v = 0; v < 8; ++v) { partial0[v] = 0.0f; partial1[v] = 0.0f; }

#pragma unroll 1
    for (int ot = 0; ot < 16; ++ot) {
        f32x8 acc0 = {0.f, 0.f, 0.f, 0.f, 0.f, 0.f, 0.f, 0.f};
        f32x8 acc1 = acc0;
        u32x8 bcur = loadB_frag(w1frag, ot, 0, lane);
#pragma unroll
        for (int c = 0; c < 8; ++c) {
            u32x8 bnext;
            if (c < 7) bnext = loadB_frag(w1frag, ot, c + 1, lane);  // prefetch
            acc0 = wmma_bf16(afrag0[c], bcur, acc0);
            acc1 = wmma_bf16(afrag1[c], bcur, acc1);
            if (c < 7) bcur = bnext;
        }
        // C/D layout: lane -> col N = col16, VGPR v -> row M = v + 8*half
        float bias = b1[ot * 16 + col16];
        float wo   = w2[ot * 16 + col16];
#pragma unroll
        for (int v = 0; v < 8; ++v) {
            float h0 = acc0[v] + bias;
            h0 = h0 > 0.0f ? h0 : 0.0f;
            partial0[v] += h0 * wo;
            float h1 = acc1[v] + bias;
            h1 = h1 > 0.0f ? h1 : 0.0f;
            partial1[v] += h1 * wo;
        }
    }

    // Reduce across the 16 N-lanes within each half-group
#pragma unroll
    for (int v = 0; v < 8; ++v) {
        float s0 = partial0[v];
        s0 += __shfl_xor(s0, 1, 32);
        s0 += __shfl_xor(s0, 2, 32);
        s0 += __shfl_xor(s0, 4, 32);
        s0 += __shfl_xor(s0, 8, 32);
        partial0[v] = s0;
        float s1 = partial1[v];
        s1 += __shfl_xor(s1, 1, 32);
        s1 += __shfl_xor(s1, 2, 32);
        s1 += __shfl_xor(s1, 4, 32);
        s1 += __shfl_xor(s1, 8, 32);
        partial1[v] = s1;
    }
    if (col16 == 0) {
        float bb = b2[0];
#pragma unroll
        for (int v = 0; v < 8; ++v) {
            int node0 = base + half * 8 + v;
            if (node0 < N) gate[node0] = sigmoidf_(partial0[v] + bb);
            int node1 = base + 16 + half * 8 + v;
            if (node1 < N) gate[node1] = sigmoidf_(partial1[v] + bb);
        }
    }
}

// ---------------------------------------------------------------------------
// Fused GRU cell + ReLU over 16-graph tiles (1 wave / block):
//   gi = xin @ w_ih^T + b_ih ; gh = h @ w_hh^T + b_hh  (WMMA bf16, f32 acc)
//   r=sig(i_r+h_r); z=sig(i_z+h_z); n=tanh(i_n + r*h_n)
//   out = relu((1-z)*n + z*h)
// w_ih/w_hh fragments have 48 output tiles: [0..15]=r, [16..31]=z, [32..47]=n.
// ---------------------------------------------------------------------------
__global__ void __launch_bounds__(32) gru_kernel(const float* __restrict__ xin,
                                                 const float* __restrict__ h,
                                                 const unsigned int* __restrict__ wihfrag,
                                                 const unsigned int* __restrict__ whhfrag,
                                                 const float* __restrict__ b_ih,
                                                 const float* __restrict__ b_hh,
                                                 float* __restrict__ out) {
    __shared__ unsigned int ldsX[16 * 128];
    __shared__ unsigned int ldsH[16 * 128];
    const int lane  = threadIdx.x;
    const int g0    = blockIdx.x * 16;
    const int half  = lane >> 4;
    const int col16 = lane & 15;

    for (int i = 0; i < 16; ++i) {
        const float4* xr = (const float4*)(xin + (size_t)(g0 + i) * H);
        const float4* hr = (const float4*)(h + (size_t)(g0 + i) * H);
#pragma unroll
        for (int j = 0; j < 2; ++j) {
            int f = lane + j * 32;
            float4 a = xr[f];
            ldsX[i * 128 + f * 2]     = pack2bf(a.x, a.y);
            ldsX[i * 128 + f * 2 + 1] = pack2bf(a.z, a.w);
            float4 b = hr[f];
            ldsH[i * 128 + f * 2]     = pack2bf(b.x, b.y);
            ldsH[i * 128 + f * 2 + 1] = pack2bf(b.z, b.w);
        }
    }
    __syncthreads();

#pragma unroll 1
    for (int ot = 0; ot < 16; ++ot) {
        f32x8 a_ir = {0.f,0.f,0.f,0.f,0.f,0.f,0.f,0.f};
        f32x8 a_iz = a_ir, a_in = a_ir, a_hr = a_ir, a_hz = a_ir, a_hn = a_ir;
#pragma unroll
        for (int c = 0; c < 8; ++c) {
            u32x8 ax = loadA_lds(ldsX, col16, half, c);
            u32x8 ah = loadA_lds(ldsH, col16, half, c);
            a_ir = wmma_bf16(ax, loadB_frag(wihfrag, ot,      c, lane), a_ir);
            a_iz = wmma_bf16(ax, loadB_frag(wihfrag, ot + 16, c, lane), a_iz);
            a_in = wmma_bf16(ax, loadB_frag(wihfrag, ot + 32, c, lane), a_in);
            a_hr = wmma_bf16(ah, loadB_frag(whhfrag, ot,      c, lane), a_hr);
            a_hz = wmma_bf16(ah, loadB_frag(whhfrag, ot + 16, c, lane), a_hz);
            a_hn = wmma_bf16(ah, loadB_frag(whhfrag, ot + 32, c, lane), a_hn);
        }
        int col = ot * 16 + col16;
        float bir = b_ih[col], biz = b_ih[H + col], bin = b_ih[2 * H + col];
        float bhr = b_hh[col], bhz = b_hh[H + col], bhn = b_hh[2 * H + col];
#pragma unroll
        for (int v = 0; v < 8; ++v) {
            int row = g0 + half * 8 + v;
            float r = sigmoidf_((a_ir[v] + bir) + (a_hr[v] + bhr));
            float z = sigmoidf_((a_iz[v] + biz) + (a_hz[v] + bhz));
            float n = tanhf((a_in[v] + bin) + r * (a_hn[v] + bhn));
            float hold = h[(size_t)row * H + col];
            float hnew = (1.0f - z) * n + z * hold;
            out[(size_t)row * H + col] = hnew > 0.0f ? hnew : 0.0f;
        }
    }
}

// ---------------------------------------------------------------------------
extern "C" void kernel_launch(void* const* d_in, const int* in_sizes, int n_in,
                              void* d_out, int out_size, void* d_ws, size_t ws_size,
                              hipStream_t stream) {
    const float* x    = (const float*)d_in[0];
    const float* gw1  = (const float*)d_in[1];
    const float* gb1  = (const float*)d_in[2];
    const float* gw2  = (const float*)d_in[3];
    const float* gb2  = (const float*)d_in[4];
    const float* wih  = (const float*)d_in[5];
    const float* whh  = (const float*)d_in[6];
    const float* bih  = (const float*)d_in[7];
    const float* bhh  = (const float*)d_in[8];
    const int*   batch = (const int*)d_in[9];
    // d_in[10] = num_graphs (device scalar) -- fixed at GNUM=4096 per reference.

    const int N = in_sizes[0] / H;
    float* out = (float*)d_out;

    // Carve workspace
    char*  ws  = (char*)d_ws;
    size_t off = 0;
    auto carve = [&](size_t bytes) -> void* {
        void* p = ws + off;
        off += (bytes + 255) & ~(size_t)255;
        return p;
    };
    float* gate  = (float*)carve((size_t)N * sizeof(float));
    float* reprA = (float*)carve((size_t)GNUM * H * sizeof(float));
    float* reprB = (float*)carve((size_t)GNUM * H * sizeof(float));
    float* reprC = (float*)carve((size_t)GNUM * H * sizeof(float));
    unsigned int* w1f  = (unsigned int*)carve((size_t)16 * 2048 * 4);
    unsigned int* wihf = (unsigned int*)carve((size_t)48 * 2048 * 4);
    unsigned int* whhf = (unsigned int*)carve((size_t)48 * 2048 * 4);
    (void)ws_size; (void)n_in; (void)out_size;

    // Pre-pack weights into WMMA B fragments (bf16)
    make_bfrag_kernel<<<(16 * 2048 + 255) / 256, 256, 0, stream>>>(gw1, w1f, 16);
    make_bfrag_kernel<<<(48 * 2048 + 255) / 256, 256, 0, stream>>>(wih, wihf, 48);
    make_bfrag_kernel<<<(48 * 2048 + 255) / 256, 256, 0, stream>>>(whh, whhf, 48);

    const int blocks = (N + 31) / 32;   // two 16-node tiles per wave

    // Initial mean pool: reprA = mean_pool(x)
    pool_kernel<<<GNUM, 256, 0, stream>>>(x, nullptr, batch, reprA, N, 0);

    // Timestep 0
    gate_kernel<<<blocks, 32, 0, stream>>>(x, reprA, batch, w1f, gb1, gw2, gb2, gate, N);
    pool_kernel<<<GNUM, 256, 0, stream>>>(x, gate, batch, reprB, N, 1);
    gru_kernel<<<GNUM / 16, 32, 0, stream>>>(reprB, reprA, wihf, whhf, bih, bhh, reprC);

    // Timestep 1
    gate_kernel<<<blocks, 32, 0, stream>>>(x, reprC, batch, w1f, gb1, gw2, gb2, gate, N);
    pool_kernel<<<GNUM, 256, 0, stream>>>(x, gate, batch, reprB, N, 1);
    gru_kernel<<<GNUM / 16, 32, 0, stream>>>(reprB, reprC, wihf, whhf, bih, bhh, out);
}
